// RadialDiffusionIntegrator_50122268344520
// MI455X (gfx1250) — matrix operand order repaired and checked by reading, not compiled, not run on hardware
//
#include <hip/hip_runtime.h>
#include <hip/hip_bf16.h>

// ---------------------------------------------------------------------------
// RadialDiffusionIntegrator for MI455X (gfx1250, wave32, WMMA)
//
// Roofline: mask = 4x4096x4096 fp32 = 256MB, read 6x (2 force() per step, 3
// steps) => 1.5GB naive. Binary mask is EXACT in f16 => convert once to a
// 128MB f16 copy that fits in the 192MB L2; the 6 GEMM passes then stream it
// from L2. HBM traffic ~ 256MB + 128MB ~= 17us at 23.3 TB/s.
//
// Precision: state (B operand) split into hi/lo f16 limbs in separate WMMA
// columns; A entries are exactly 0/1 so all products are exact and only the
// fp32 WMMA accumulation rounds -> ~fp32-accurate result.
//
// Round-2 change: branchless inner loop. B columns 4..15 read from a zeroed
// LDS plane instead of per-iteration EXEC-masked zero fills (asm showed 8
// v_dual_mov zeros + EXEC save/restore per WMMA). Unroll deepened to 4.
// ---------------------------------------------------------------------------

typedef __attribute__((ext_vector_type(16))) _Float16 v16h;
typedef __attribute__((ext_vector_type(8)))  _Float16 v8h;
typedef __attribute__((ext_vector_type(8)))  float    v8f;

#define SEQ   4096
#define BATCH 4
#define DT_C      0.1f
#define EPSILON_C 1e-8f

// --- mask fp32 -> f16 (exact for 0/1 values) ------------------------------
__global__ __launch_bounds__(256) void cvt_mask_f16(const float* __restrict__ m,
                                                    _Float16* __restrict__ mh) {
    long long i = ((long long)blockIdx.x * blockDim.x + threadIdx.x) * 4;
    float4 v = *reinterpret_cast<const float4*>(m + i);
    union { _Float16 h[4]; uint2 u; } pk;
    pk.h[0] = (_Float16)v.x; pk.h[1] = (_Float16)v.y;
    pk.h[2] = (_Float16)v.z; pk.h[3] = (_Float16)v.w;
    *reinterpret_cast<uint2*>(mh + i) = pk.u;
}

__global__ __launch_bounds__(256) void copy_state(const float* __restrict__ src,
                                                  float* __restrict__ dst, int n) {
    int i = blockIdx.x * blockDim.x + threadIdx.x;
    if (i < n) dst[i] = src[i];
}

// ---------------------------------------------------------------------------
// One force() pass fused with its RK2 epilogue.
// PHASE 1: k1 = M p - p ; psi_star = renorm(p + DT*k1, r) ; emits star,k1,r
// PHASE 2: k2 = M s - s ; p_new = renorm(p + .5*DT*(k1+k2), r) ; emits state
// Block = 128 threads = 4 waves; each block owns 16 output rows; each wave
// covers one K-quarter with 32x v_wmma_f32_16x16x32_f16.
// ---------------------------------------------------------------------------
template <int PHASE, bool F16MASK>
__global__ __launch_bounds__(128) void rk_pass(
    const _Float16* __restrict__ maskH,   // f16 mask (if F16MASK)
    const float*    __restrict__ maskF,   // f32 mask (fallback path)
    const float*    __restrict__ x,       // GEMM input state (p or psi_star)
    const float*    __restrict__ p,       // step's base state
    const float*    __restrict__ k1buf,   // PHASE 2 only
    const float*    __restrict__ rbuf,    // PHASE 2 only
    float* __restrict__ out_state,
    float* __restrict__ out_k1,           // PHASE 1 only
    float* __restrict__ out_r)            // PHASE 1 only
{
    // Limb planes: hi0, lo0, hi1, lo1, ZERO (plane 4 feeds B columns 4..15
    // so the inner loop is branchless). 5 * 4096 * 2B = 40KB.
    __shared__ _Float16 BT[5 * SEQ];
    __shared__ float    red[4][64];       // per-wave partial 16x4 tiles

    const int blk   = blockIdx.x;
    const int batch = blk >> 8;           // 256 row-tiles per batch
    const int tile  = blk & 255;
    const int row0  = tile << 4;
    const int tid   = threadIdx.x;
    const int lane  = tid & 31;
    const int wave  = tid >> 5;

    // Stage hi/lo f16 split of full state column into LDS (limb-major),
    // and zero-fill the padding plane.
    const float* xb = x + (size_t)batch * SEQ * 2;
    for (int j = tid; j < SEQ; j += 128) {
        float x0 = xb[2 * j], x1 = xb[2 * j + 1];
        _Float16 h0 = (_Float16)x0;
        _Float16 l0 = (_Float16)(x0 - (float)h0);
        _Float16 h1 = (_Float16)x1;
        _Float16 l1 = (_Float16)(x1 - (float)h1);
        BT[0 * SEQ + j] = h0;  BT[1 * SEQ + j] = l0;
        BT[2 * SEQ + j] = h1;  BT[3 * SEQ + j] = l1;
        BT[4 * SEQ + j] = (_Float16)0.0f;
    }
    __syncthreads();

    const int m      = lane & 15;         // A row within tile / B,C column
    const int hiHalf = lane >> 4;         // lane group selects K sub-range
    const _Float16* arowH = maskH + (size_t)batch * SEQ * SEQ + (size_t)(row0 + m) * SEQ;
    const float*    arowF = maskF + (size_t)batch * SEQ * SEQ + (size_t)(row0 + m) * SEQ;

    // B base pointer: columns >= 4 read the zero plane (branchless loop).
    const int bplane = (m < 4) ? m : 4;
    const _Float16* bbase = &BT[bplane * SEQ + hiHalf * 8];

    v8f c = {};
    const int kstart = wave * (SEQ / 4);

    #pragma unroll 4
    for (int kk = 0; kk < SEQ / 4; kk += 32) {
        const int k0 = kstart + kk;
        // A 16x32 f16 fragment: lanes 0-15 hold K {0-7,16-23}, lanes 16-31
        // hold K {8-15,24-31} (ISA 7.12.2) -> two 16B loads at ka, ka+16.
        const int ka = k0 + hiHalf * 8;
        v16h A;
        if (F16MASK) {
            v8h a0 = *reinterpret_cast<const v8h*>(arowH + ka);
            v8h a1 = *reinterpret_cast<const v8h*>(arowH + ka + 16);
            A = __builtin_shufflevector(a0, a1, 0,1,2,3,4,5,6,7,8,9,10,11,12,13,14,15);
        } else {
            const float* af0 = arowF + ka;
            const float* af1 = arowF + ka + 16;
            #pragma unroll
            for (int t = 0; t < 8; ++t) { A[t] = (_Float16)af0[t]; A[8 + t] = (_Float16)af1[t]; }
        }
        // B 32x16 f16 fragment: lanes 0-15 hold K {0-7,16-23}, lanes 16-31
        // hold K {8-15,24-31} for column m (same order as A).
        const _Float16* bp = bbase + k0;
        v8h b0 = *reinterpret_cast<const v8h*>(bp);
        v8h b1 = *reinterpret_cast<const v8h*>(bp + 16);
        v16h Bm = __builtin_shufflevector(b0, b1, 0,1,2,3,4,5,6,7,8,9,10,11,12,13,14,15);

        c = __builtin_amdgcn_wmma_f32_16x16x32_f16(false, A, false, Bm,
                                                   (short)0, c, false, false);
    }

    // Dump meaningful columns (0..3) of this wave's partial 16x16 tile.
    // C layout: lanes 0-15 -> M = v ; lanes 16-31 -> M = v + 8 (ISA 7.12.2).
    if (m < 4) {
        const int mOff = hiHalf * 8;
        #pragma unroll
        for (int v = 0; v < 8; ++v)
            red[wave][(mOff + v) * 4 + m] = c[v];
    }
    __syncthreads();

    // Cross-wave reduction + hi/lo limb combine + RK2 epilogue (row-local).
    if (tid < 16) {
        const int row = tid;
        float mx0 = 0.f, mx1 = 0.f;
        #pragma unroll
        for (int w = 0; w < 4; ++w) {
            mx0 += red[w][row * 4 + 0] + red[w][row * 4 + 1];
            mx1 += red[w][row * 4 + 2] + red[w][row * 4 + 3];
        }
        const int g = batch * SEQ + row0 + row;
        if (PHASE == 1) {
            float p0 = p[2 * g], p1 = p[2 * g + 1];
            float k10 = mx0 - p0, k11 = mx1 - p1;           // force(p)
            float r = sqrtf(p0 * p0 + p1 * p1);
            float s0 = p0 + DT_C * k10, s1 = p1 + DT_C * k11;
            float sn = sqrtf(s0 * s0 + s1 * s1);
            float sc = r / (sn + EPSILON_C);
            out_state[2 * g] = s0 * sc;  out_state[2 * g + 1] = s1 * sc;
            out_k1[2 * g]    = k10;      out_k1[2 * g + 1]    = k11;
            out_r[g] = r;
        } else {
            float s0 = x[2 * g], s1 = x[2 * g + 1];
            float k20 = mx0 - s0, k21 = mx1 - s1;           // force(psi_star)
            float p0 = p[2 * g], p1 = p[2 * g + 1];
            float k10 = k1buf[2 * g], k11 = k1buf[2 * g + 1];
            float pn0 = p0 + 0.5f * DT_C * (k10 + k20);
            float pn1 = p1 + 0.5f * DT_C * (k11 + k21);
            float r = rbuf[g];
            float nn = sqrtf(pn0 * pn0 + pn1 * pn1);
            float sc = r / (nn + EPSILON_C);
            out_state[2 * g] = pn0 * sc;  out_state[2 * g + 1] = pn1 * sc;
        }
    }
}

extern "C" void kernel_launch(void* const* d_in, const int* in_sizes, int n_in,
                              void* d_out, int out_size, void* d_ws, size_t ws_size,
                              hipStream_t stream) {
    const float* psi  = (const float*)d_in[0];
    const float* mask = (const float*)d_in[1];
    float* out = (float*)d_out;

    const size_t maskElems  = (size_t)BATCH * SEQ * SEQ;      // 67,108,864
    const size_t maskHBytes = maskElems * sizeof(_Float16);   // 128 MB
    const size_t stBytes    = (size_t)BATCH * SEQ * 2 * sizeof(float); // 128 KB
    const size_t rBytes     = (size_t)BATCH * SEQ * sizeof(float);     // 64 KB

    char* ws = (char*)d_ws;
    const bool useF16 = ws_size >= maskHBytes + 3 * stBytes + rBytes;

    size_t off = useF16 ? maskHBytes : 0;
    _Float16* maskH = (_Float16*)ws;
    float* p    = (float*)(ws + off); off += stBytes;
    float* k1   = (float*)(ws + off); off += stBytes;
    float* star = (float*)(ws + off); off += stBytes;
    float* rbuf = (float*)(ws + off);

    if (useF16) {
        // 67,108,864 elems / (256 thr * 4 per thr) = 65536 blocks, exact.
        cvt_mask_f16<<<(unsigned)(maskElems / (256 * 4)), 256, 0, stream>>>(mask, maskH);
    }
    const int nstate = BATCH * SEQ * 2;
    copy_state<<<(nstate + 255) / 256, 256, 0, stream>>>(psi, p, nstate);

    dim3 grid(BATCH * (SEQ / 16));   // 1024 blocks, 128 threads (4 waves) each
    for (int step = 0; step < 3; ++step) {
        float* dst = (step == 2) ? out : p;
        if (useF16) {
            rk_pass<1, true ><<<grid, 128, 0, stream>>>(maskH, mask, p,    p, nullptr, nullptr, star, k1, rbuf);
            rk_pass<2, true ><<<grid, 128, 0, stream>>>(maskH, mask, star, p, k1,      rbuf,    dst, nullptr, nullptr);
        } else {
            rk_pass<1, false><<<grid, 128, 0, stream>>>(maskH, mask, p,    p, nullptr, nullptr, star, k1, rbuf);
            rk_pass<2, false><<<grid, 128, 0, stream>>>(maskH, mask, star, p, k1,      rbuf,    dst, nullptr, nullptr);
        }
    }
}
